// Twister_19232863551565
// MI455X (gfx1250) — matrix operand order repaired and checked by reading, not compiled, not run on hardware
//
#include <hip/hip_runtime.h>
#include <hip/hip_bf16.h>
#include <stdint.h>

typedef __attribute__((ext_vector_type(16))) _Float16 v16h;
typedef __attribute__((ext_vector_type(8)))  float    v8f;

#define DEV __device__ __forceinline__

DEV int   imin(int a, int b)  { return a < b ? a : b; }
DEV float silu_f(float x)     { return x / (1.f + __expf(-x)); }
DEV float softplus_f(float x) { return (x > 20.f) ? x : logf(1.f + __expf(x)); }

DEV void bil_coef(int g, int n_in, int n_out, int& i0, int& i1, float& f) {
  float src = (g + 0.5f) * ((float)n_in / (float)n_out) - 0.5f;
  src = fminf(fmaxf(src, 0.f), (float)(n_in - 1));
  int a = (int)floorf(src);
  int b = a + 1; if (b > n_in - 1) b = n_in - 1;
  i0 = a; i1 = b; f = src - (float)a;
}

// ---------------------------------------------------------------------------
// WMMA GEMM tile worker: one wave32 computes a 16(M) x 32(N) output tile as
// two 16x16 WMMAs sharing the A operand. f32 in/out, f16 multiply, f32 acc.
// EDGE=false: no guards (K%32==0, full M tile) -> clean vectorized loads.
// EDGE=true : clamped-index + select zeroing (branch-free, no exec waves).
// Lane layouts used here (A/B k contiguous per half, not the ISA table order):
//   A 16x32 f16 : m=lane&15, half=lane>>4; vgpr v: K=half*16+2v
//   B 32x16 f16 : n=lane&15, half=lane>>4; vgpr v: K=half*16+2v
//   C 16x16 f32 : n=lane&15; vgpr r: M=r+8*half
// ---------------------------------------------------------------------------
template <bool EDGE>
DEV void gemm_tile(const float* __restrict__ Ab, int lda,
                   const float* __restrict__ Bb, int ldb,
                   float* __restrict__ Cb, int ldc,
                   int M, int K, int m0, int n0, int lane)
{
  const int half = lane >> 4;
  const int ln   = lane & 15;
  const int mr   = m0 + ln;
  const int mrc  = EDGE ? imin(mr, M - 1) : mr;
  const int nc0  = n0 + ln;
  const int nc1  = nc0 + 16;
  const float* Arow = Ab + (long)mrc * lda;

  v8f acc0 = {}, acc1 = {};
  for (int k0 = 0; k0 < K; k0 += 32) {
    v16h av, bv0, bv1;
#pragma unroll
    for (int v = 0; v < 8; ++v) {
      const int ka = k0 + (half << 4) + (v << 1);
      float a0, a1;
      if (EDGE) {
        a0 = Arow[imin(ka,     K - 1)];
        a1 = Arow[imin(ka + 1, K - 1)];
        a0 = (mr < M && ka     < K) ? a0 : 0.f;
        a1 = (mr < M && ka + 1 < K) ? a1 : 0.f;
      } else {
        const float2 ap = *(const float2*)(Arow + ka);   // global_load_b64
        a0 = ap.x; a1 = ap.y;
      }
      av[2*v]   = (_Float16)a0;
      av[2*v+1] = (_Float16)a1;

      const int kb = k0 + (half << 4) + (v << 1);
      float b00, b10, b01, b11;
      if (EDGE) {
        const long r0 = (long)imin(kb,     K - 1) * ldb;
        const long r1 = (long)imin(kb + 1, K - 1) * ldb;
        b00 = Bb[r0 + nc0]; b01 = Bb[r0 + nc1];
        b10 = Bb[r1 + nc0]; b11 = Bb[r1 + nc1];
        if (kb     >= K) { b00 = 0.f; b01 = 0.f; }
        if (kb + 1 >= K) { b10 = 0.f; b11 = 0.f; }
      } else {
        const long r0 = (long)kb * ldb, r1 = r0 + ldb;
        b00 = Bb[r0 + nc0]; b01 = Bb[r0 + nc1];
        b10 = Bb[r1 + nc0]; b11 = Bb[r1 + nc1];
      }
      bv0[2*v]   = (_Float16)b00;
      bv0[2*v+1] = (_Float16)b10;
      bv1[2*v]   = (_Float16)b01;
      bv1[2*v+1] = (_Float16)b11;
    }
    acc0 = __builtin_amdgcn_wmma_f32_16x16x32_f16(false, av, false, bv0,
                                                  (short)0, acc0, false, false);
    acc1 = __builtin_amdgcn_wmma_f32_16x16x32_f16(false, av, false, bv1,
                                                  (short)0, acc1, false, false);
  }
#pragma unroll
  for (int r = 0; r < 8; ++r) {
    const int m = m0 + r + (half << 3);
    if (!EDGE || m < M) {
      Cb[(long)m * ldc + nc0] = acc0[r];
      Cb[(long)m * ldc + nc1] = acc1[r];
    }
  }
}

// Batched GEMM: C[bz] (MxN) = A[bz % a_mod] (MxK) * B[bz] (KxN), row-major.
// N must be a multiple of 32.
__global__ void k_wmma_gemm(const float* __restrict__ A, long a_stride, int a_mod, int lda,
                            const float* __restrict__ Bm, long b_stride, int ldb,
                            float* __restrict__ C, long c_stride, int ldc,
                            int M, int N, int K, int batches)
{
  const int wave = (int)(threadIdx.x >> 5);
  const int lane = (int)(threadIdx.x & 31);
  const int tiles_n = N >> 5;
  const int tiles_m = (M + 15) >> 4;
  const long tpb = (long)tiles_m * tiles_n;
  const long wg = (long)blockIdx.x * (blockDim.x >> 5) + wave;
  const long bz = wg / tpb;
  if (bz >= batches) return;
  const int t  = (int)(wg % tpb);
  const int tm = t / tiles_n, tn = t % tiles_n;

  const float* Ab = A  + (long)(bz % a_mod) * a_stride;
  const float* Bb = Bm + bz * b_stride;
  float*       Cb = C  + bz * c_stride;
  const int m0 = tm << 4, n0 = tn << 5;

  if ((m0 + 16 > M) || (K & 31))
    gemm_tile<true >(Ab, lda, Bb, ldb, Cb, ldc, M, K, m0, n0, lane);
  else
    gemm_tile<false>(Ab, lda, Bb, ldb, Cb, ldc, M, K, m0, n0, lane);
}

// ---------------------------------------------------------------------------
// Stage-1 front end: x1[b, p, c] = bilinear16(concat(img,gc,lc)[b, c])[p]
// ---------------------------------------------------------------------------
__global__ void k_resize_down(const float* __restrict__ img,
                              const float* __restrict__ gc,
                              const float* __restrict__ lc,
                              float* __restrict__ x1)
{
  long i = (long)blockIdx.x * blockDim.x + threadIdx.x;
  const long total = 4L * 256 * 288;
  if (i >= total) return;
  int c = (int)(i % 288);
  int p = (int)((i / 288) % 256);
  int b = (int)(i / (288L * 256));
  int gh = p >> 4, gw = p & 15;
  int h0, h1, w0, w1; float fh, fw;
  bil_coef(gh, 64, 16, h0, h1, fh);
  bil_coef(gw, 64, 16, w0, w1, fw);
  const float* s;
  if (c < 96)       s = img + ((long)b * 96 + c)        * 4096;
  else if (c < 192) s = gc  + ((long)b * 96 + (c - 96)) * 4096;
  else              s = lc  + ((long)b * 96 + (c - 192))* 4096;
  float v00 = s[h0*64+w0], v01 = s[h0*64+w1], v10 = s[h1*64+w0], v11 = s[h1*64+w1];
  x1[i] = (v00*(1.f-fw)+v01*fw)*(1.f-fh) + (v10*(1.f-fw)+v11*fw)*fh;
}

// x2[b,c,h,w] = bilinear64(o1 as (B,288,16,16))[h,w] + concat[b,c,h,w]
__global__ void k_upsample_add(const float* __restrict__ o1,
                               const float* __restrict__ img,
                               const float* __restrict__ gc,
                               const float* __restrict__ lc,
                               float* __restrict__ x2)
{
  long i = (long)blockIdx.x * blockDim.x + threadIdx.x;
  const long total = 4L * 288 * 4096;
  if (i >= total) return;
  int hw = (int)(i % 4096);
  int c  = (int)((i / 4096) % 288);
  int b  = (int)(i / (4096L * 288));
  int h = hw >> 6, w = hw & 63;
  int h0, h1, w0, w1; float fh, fw;
  bil_coef(h, 16, 64, h0, h1, fh);
  bil_coef(w, 16, 64, w0, w1, fw);
  long base = (long)b * 256;
  float g00 = o1[(base + h0*16 + w0) * 288 + c];
  float g01 = o1[(base + h0*16 + w1) * 288 + c];
  float g10 = o1[(base + h1*16 + w0) * 288 + c];
  float g11 = o1[(base + h1*16 + w1) * 288 + c];
  float v = (g00*(1.f-fw)+g01*fw)*(1.f-fh) + (g10*(1.f-fw)+g11*fw)*fh;
  const float* s;
  if (c < 96)       s = img + ((long)b * 96 + c)        * 4096;
  else if (c < 192) s = gc  + ((long)b * 96 + (c - 96)) * 4096;
  else              s = lc  + ((long)b * 96 + (c - 192))* 4096;
  x2[i] = v + s[hw];
}

// Depthwise 3x3 SAME conv + bias + SiLU on the xc half of xz (B, 2D, L)
__global__ void k_conv_dw(const float* __restrict__ xz,
                          const float* __restrict__ cw,
                          const float* __restrict__ cb,
                          float* __restrict__ xc,
                          int B, int D, int H, int W, int twoD)
{
  int L = H * W;
  long i = (long)blockIdx.x * blockDim.x + threadIdx.x;
  long total = (long)B * D * L;
  if (i >= total) return;
  int l = (int)(i % L);
  int d = (int)((i / L) % D);
  int b = (int)(i / ((long)L * D));
  int h = l / W, w = l % W;
  const float* src = xz + ((long)b * twoD + d) * L;
  const float* wk  = cw + (long)d * 9;
  float acc = cb[d];
#pragma unroll
  for (int dy = -1; dy <= 1; ++dy) {
    int hh = h + dy; if (hh < 0 || hh >= H) continue;
#pragma unroll
    for (int dx = -1; dx <= 1; ++dx) {
      int ww = w + dx; if (ww < 0 || ww >= W) continue;
      acc += wk[(dy+1)*3 + (dx+1)] * src[hh * W + ww];
    }
  }
  xc[((long)b * D + d) * L + l] = silu_f(acc);
}

// xs (B,4,D,L): k0=row-major, k1=transposed, k2=reversed k0, k3=reversed k1
__global__ void k_build_xs(const float* __restrict__ xc, float* __restrict__ xs,
                           int B, int D, int H, int W)
{
  int L = H * W;
  long i = (long)blockIdx.x * blockDim.x + threadIdx.x;
  long total = (long)B * D * L;
  if (i >= total) return;
  int l = (int)(i % L);
  int d = (int)((i / L) % D);
  int b = (int)(i / ((long)L * D));
  int h = l / W, w = l % W;
  int lt = w * H + h;
  float v = xc[((long)b * D + d) * L + l];
  long kst = (long)D * L;
  long s0  = ((long)b * 4 * D + d) * (long)L;
  xs[s0 + l]                  = v;
  xs[s0 + kst   + lt]         = v;
  xs[s0 + 2*kst + (L-1-l)]    = v;
  xs[s0 + 3*kst + (L-1-lt)]   = v;
}

// Selective scan, one thread per (b,k,d). N=16 state in registers.
// dtm holds raw dt_proj output on entry, receives ys on exit (in-place).
__global__ void k_scan(const float* __restrict__ xs,
                       float* __restrict__ dtm,
                       const float* __restrict__ xdbl,
                       const float* __restrict__ dtb,
                       const float* __restrict__ Alog,
                       const float* __restrict__ Ds,
                       int B, int D, int L, int R, int Cx)
{
  int i = blockIdx.x * blockDim.x + threadIdx.x;
  int total = B * 4 * D;
  if (i >= total) return;
  int d = i % D;
  int k = (i / D) & 3;
  int b = i / (4 * D);

  float A[16];
#pragma unroll
  for (int n = 0; n < 16; ++n)
    A[n] = -__expf(Alog[((long)(k * D) + d) * 16 + n]);
  float bias = dtb[k * D + d];
  float Dv   = Ds[k * D + d];

  const float* Bp = xdbl + (((long)(b*4 + k)) * Cx + R) * (long)L;
  const float* Cp = Bp + 16L * L;
  const float* uu = xs  + (((long)(b*4 + k)) * D + d) * (long)L;
  float*       up = dtm + (((long)(b*4 + k)) * D + d) * (long)L;

  float h[16];
#pragma unroll
  for (int n = 0; n < 16; ++n) h[n] = 0.f;

  for (int l = 0; l < L; ++l) {
    float dtv = softplus_f(up[l] + bias);
    float u   = uu[l];
    float du  = dtv * u;
    float y   = 0.f;
#pragma unroll
    for (int n = 0; n < 16; ++n) {
      h[n] = h[n] * __expf(dtv * A[n]) + du * Bp[(long)n * L + l];
      y   += h[n] * Cp[(long)n * L + l];
    }
    up[l] = y + u * Dv;
  }
}

// y[b,d,l] = ys0[l] + ys2[L-1-l] + ys1[m] + ys3[L-1-m],  m = w*H + h
__global__ void k_combine(const float* __restrict__ ys, float* __restrict__ y,
                          int B, int D, int H, int W)
{
  int L = H * W;
  long i = (long)blockIdx.x * blockDim.x + threadIdx.x;
  long total = (long)B * D * L;
  if (i >= total) return;
  int l = (int)(i % L);
  int d = (int)((i / L) % D);
  int b = (int)(i / ((long)L * D));
  int h = l / W, w = l % W;
  int m = w * H + h;
  long kst = (long)D * L;
  long s0  = ((long)b * 4 * D + d) * (long)L;
  float v = ys[s0 + l] + ys[s0 + 2*kst + (L-1-l)]
          + ys[s0 + kst + m] + ys[s0 + 3*kst + (L-1-m)];
  y[((long)b * D + d) * L + l] = v;
}

// Channel layernorm at each (b,l) + gate with silu(z); in-place on y.
// One wave32 per (b,l); shuffle reduction (warpSize==32 on gfx1250).
__global__ void k_norm_gate(float* __restrict__ y, const float* __restrict__ xz,
                            const float* __restrict__ nw, const float* __restrict__ nb,
                            int B, int D, int L, int twoD)
{
  long gw = ((long)blockIdx.x * blockDim.x + threadIdx.x) >> 5;
  int lane = (int)(threadIdx.x & 31);
  if (gw >= (long)B * L) return;
  int b = (int)(gw / L), l = (int)(gw % L);
  float* yb = y + (long)b * D * L + l;
  float s = 0.f, ss = 0.f;
  for (int d = lane; d < D; d += 32) {
    float v = yb[(long)d * L];
    s += v; ss += v * v;
  }
#pragma unroll
  for (int m = 16; m > 0; m >>= 1) {
    s  += __shfl_xor(s,  m, 32);
    ss += __shfl_xor(ss, m, 32);
  }
  float mu  = s / (float)D;
  float var = ss / (float)D - mu * mu;
  float rs  = rsqrtf(var + 1e-5f);
  const float* zb = xz + ((long)b * twoD + D) * L + l;
  for (int d = lane; d < D; d += 32) {
    float v = (yb[(long)d * L] - mu) * rs * nw[d] + nb[d];
    float z = zb[(long)d * L];
    yb[(long)d * L] = v * silu_f(z);
  }
}

// ---------------------------------------------------------------------------
// Host orchestration
// ---------------------------------------------------------------------------
struct SSParams {
  const float *in_w, *conv_w, *conv_b, *xproj_w, *dt_w, *dt_b;
  const float *A_logs, *Ds, *nw, *nb, *out_w;
};

static inline void launch_gemm(hipStream_t st,
                               const float* A, long a_stride, int a_mod, int lda,
                               const float* B, long b_stride, int ldb,
                               float* C, long c_stride, int ldc,
                               int M, int N, int K, int batches)
{
  int tiles_m = (M + 15) / 16, tiles_n = N / 32;   // 16x32 tile per wave
  long waves  = (long)batches * tiles_m * tiles_n;
  const int wpb = 4;                               // 4 wave32 per block
  long blocks = (waves + wpb - 1) / wpb;
  k_wmma_gemm<<<dim3((unsigned)blocks), dim3(wpb * 32), 0, st>>>(
      A, a_stride, a_mod, lda, B, b_stride, ldb, C, c_stride, ldc, M, N, K, batches);
}

static void run_ss2d(hipStream_t st, const SSParams& p,
                     const float* x, float* out,
                     float* xz, float* xc, float* xs, float* xdbl, float* dtm,
                     int Bn, int Cin, int H, int W, int D, int R, int Cout)
{
  const int L = H * W, twoD = 2 * D, Cx = R + 32;   // N=16 -> Cx = R + 2N
  // in_proj: xz (B, 2D, L) = W_in (2D x Cin) * x (B, Cin, L)
  launch_gemm(st, p.in_w, 0, 1, Cin,  x, (long)Cin * L, L,
              xz, (long)twoD * L, L,  twoD, L, Cin, Bn);
  // depthwise conv + SiLU on xc half
  { long n = (long)Bn * D * L;
    k_conv_dw<<<(unsigned)((n + 255) / 256), 256, 0, st>>>(xz, p.conv_w, p.conv_b,
                                                           xc, Bn, D, H, W, twoD); }
  // build 4-direction sequences
  { long n = (long)Bn * D * L;
    k_build_xs<<<(unsigned)((n + 255) / 256), 256, 0, st>>>(xc, xs, Bn, D, H, W); }
  // x_proj: xdbl (B,4,Cx,L) = x_proj_w[k] (Cx x D) * xs[b,k] (D x L)
  launch_gemm(st, p.xproj_w, (long)Cx * D, 4, D,  xs, (long)D * L, L,
              xdbl, (long)Cx * L, L,  Cx, L, D, Bn * 4);
  // dt_proj: dtm (B,4,D,L) = dt_proj_w[k] (D x R) * xdbl rows [0,R)
  launch_gemm(st, p.dt_w, (long)D * R, 4, R,  xdbl, (long)Cx * L, L,
              dtm, (long)D * L, L,  D, L, R, Bn * 4);
  // selective scan (softplus+bias fused); ys written in-place over dtm
  { long n = (long)Bn * 4 * D;
    k_scan<<<(unsigned)((n + 255) / 256), 256, 0, st>>>(xs, dtm, xdbl, p.dt_b,
                                                        p.A_logs, p.Ds, Bn, D, L, R, Cx); }
  // merge directions -> y (reuse xc buffer)
  { long n = (long)Bn * D * L;
    k_combine<<<(unsigned)((n + 255) / 256), 256, 0, st>>>(dtm, xc, Bn, D, H, W); }
  // layernorm over channels + silu(z) gate, in place
  { long waves = (long)Bn * L;
    k_norm_gate<<<(unsigned)((waves * 32 + 255) / 256), 256, 0, st>>>(xc, xz, p.nw, p.nb,
                                                                      Bn, D, L, twoD); }
  // out_proj: out (B, Cout, L) = W_out (Cout x D) * y (B, D, L)
  launch_gemm(st, p.out_w, 0, 1, D,  xc, (long)D * L, L,
              out, (long)Cout * L, L,  Cout, L, D, Bn);
}

extern "C" void kernel_launch(void* const* d_in, const int* in_sizes, int n_in,
                              void* d_out, int out_size, void* d_ws, size_t ws_size,
                              hipStream_t stream)
{
  (void)in_sizes; (void)n_in; (void)out_size; (void)ws_size;
  const float* img = (const float*)d_in[0];
  const float* gcv = (const float*)d_in[1];
  const float* lcv = (const float*)d_in[2];
  SSParams p1 = { (const float*)d_in[3],  (const float*)d_in[4],  (const float*)d_in[5],
                  (const float*)d_in[6],  (const float*)d_in[7],  (const float*)d_in[8],
                  (const float*)d_in[9],  (const float*)d_in[10], (const float*)d_in[11],
                  (const float*)d_in[12], (const float*)d_in[13] };
  SSParams p2 = { (const float*)d_in[14], (const float*)d_in[15], (const float*)d_in[16],
                  (const float*)d_in[17], (const float*)d_in[18], (const float*)d_in[19],
                  (const float*)d_in[20], (const float*)d_in[21], (const float*)d_in[22],
                  (const float*)d_in[23], (const float*)d_in[24] };

  // Workspace arena (fp32 elements); sized for the larger stage-2 shapes.
  float* w    = (float*)d_ws;
  float* o1   = w;                       // 4*256*288        = 294912
  float* xbuf = o1   + 294912;           // 4*288*4096       = 4718592 (x1 / x2)
  float* xz   = xbuf + 4718592;          // 4*384*4096       = 6291456
  float* xc   = xz   + 6291456;          // 4*192*4096       = 3145728 (also y)
  float* xs   = xc   + 3145728;          // 4*4*192*4096     = 12582912
  float* xdbl = xs   + 12582912;         // 4*4*48*4096      = 3145728 (max of stages)
  float* dtm  = xdbl + 3145728;          // 4*4*192*4096     = 12582912 (also ys)
  // total = 42,762,240 floats = ~171 MB of d_ws

  // ---- Stage 1: resize to 16x16, transpose into (B, 256, 288) "image" ----
  { long n = 4L * 256 * 288;
    k_resize_down<<<(unsigned)((n + 255) / 256), 256, 0, stream>>>(img, gcv, lcv, xbuf); }
  // SS2D #1: Cin=256, H=1, W=288, D=512, R=16, Cout=256
  run_ss2d(stream, p1, xbuf, o1, xz, xc, xs, xdbl, dtm, 4, 256, 1, 288, 512, 16, 256);

  // ---- Between stages: upsample 16x16 -> 64x64 and add residual ----
  { long n = 4L * 288 * 4096;
    k_upsample_add<<<(unsigned)((n + 255) / 256), 256, 0, stream>>>(o1, img, gcv, lcv, xbuf); }

  // SS2D #2: Cin=288, H=64, W=64, D=192, R=6, Cout=96 -> writes d_out directly
  run_ss2d(stream, p2, xbuf, (float*)d_out, xz, xc, xs, xdbl, dtm, 4, 288, 64, 64, 192, 6, 96);
}
